// LocalSelfAttention_54984171323856
// MI455X (gfx1250) — matrix operand order repaired and checked by reading, not compiled, not run on hardware
//
#include <hip/hip_runtime.h>
#include <hip/hip_bf16.h>

typedef __attribute__((ext_vector_type(16))) __bf16 v16bf;
typedef __attribute__((ext_vector_type(8)))  float  v8f;

union FragAB { v16bf v; unsigned int u[8]; };

__device__ __forceinline__ unsigned short f2bf(float x) {
  unsigned int u = __float_as_uint(x);
  u += 0x7FFFu + ((u >> 16) & 1u);          // round-to-nearest-even
  return (unsigned short)(u >> 16);
}
__device__ __forceinline__ unsigned int pack2bf(float lo, float hi) {
  return (unsigned int)f2bf(lo) | ((unsigned int)f2bf(hi) << 16);
}

// ---- WMMA fragment loads (contiguous k = 16*(lane/16) + s, wave32) -----------------------

// A: 16x32 bf16 tile from row-major buf[(m0+m)*ld + k].
__device__ __forceinline__ v16bf loadA(const unsigned short* buf, int ld,
                                       int m0, int k0, int lane) {
  FragAB f;
  const int g = lane >> 4, m = lane & 15;
  const unsigned short* row = buf + (m0 + m) * ld + k0 + 16 * g;
#pragma unroll
  for (int i = 0; i < 8; ++i) {
    const int kb = 2 * i;
    f.u[i] = *reinterpret_cast<const unsigned int*>(row + kb);
  }
  return f.v;
}

// B: 32x16 bf16 tile; element (k, n) stored K-contiguous at bufT[(n0+n)*ld + k].
__device__ __forceinline__ v16bf loadB(const unsigned short* bufT, int ld,
                                       int n0, int k0, int lane) {
  FragAB f;
  const int g = lane >> 4, n = lane & 15;
  const unsigned short* col = bufT + (n0 + n) * ld + k0 + 16 * g;
#pragma unroll
  for (int i = 0; i < 8; ++i)
    f.u[i] = *reinterpret_cast<const unsigned int*>(col + 2 * i);
  return f.v;
}

__device__ __forceinline__ v8f bfmma(v16bf a, v16bf b, v8f c) {
  return __builtin_amdgcn_wmma_f32_16x16x32_bf16(false, a, false, b,
                                                 (short)0, c, false, false);
}

// ---- prep: transpose + convert weights to bf16 ------------------------------
__global__ void prep_weights_kernel(const float* __restrict__ w_qkv,
                                    const float* __restrict__ w_proj,
                                    unsigned short* __restrict__ wqkvT,
                                    unsigned short* __restrict__ wprojT) {
  const int idx = blockIdx.x * 256 + threadIdx.x;
  if (idx < 1152 * 384) {               // wqkvT[j*384+c] = bf16(w_qkv[c*1152+j])
    const int j = idx / 384, c = idx - j * 384;
    wqkvT[idx] = f2bf(w_qkv[c * 1152 + j]);
  } else if (idx < 1152 * 384 + 384 * 384) {
    const int t = idx - 1152 * 384;
    const int j = t / 384, c = t - j * 384;
    wprojT[t] = f2bf(w_proj[c * 384 + j]);
  }
}

// ---- fused window attention -------------------------------------------------
// LDS leading dims chosen so row stride in dwords is ODD -> conflict-free
// fragment loads (LDX=390 -> 195 dw, LDVT/LDP=70 -> 35 dw).
#define LDX   390
#define LDVT   70
#define LDSS   66
#define LDP    70
// byte offsets inside dynamic LDS (total 229376 B <= 320 KB/WG)
#define OFF_Q   49920
#define OFF_K   99840
#define OFF_VT 149760
#define OFF_S  203520
#define OFF_P  220416

#define NTHREADS 512
#define NWAVES   16

__global__ __launch_bounds__(NTHREADS, 1)
void win_attn_kernel(const float* __restrict__ x,
                     const float* __restrict__ b_qkv,
                     const float* __restrict__ b_proj,
                     const unsigned short* __restrict__ wqkvT,
                     const unsigned short* __restrict__ wprojT,
                     float* __restrict__ out) {
  extern __shared__ unsigned char smem[];
  unsigned short* sX  = (unsigned short*)(smem);            // 64 x LDX (later O)
  unsigned short* sQ  = (unsigned short*)(smem + OFF_Q);    // 64 x LDX
  unsigned short* sK  = (unsigned short*)(smem + OFF_K);    // 64 x LDX
  unsigned short* sVT = (unsigned short*)(smem + OFF_VT);   // 384 x LDVT (V^T)
  float*          sS  = (float*)        (smem + OFF_S);     // 64 x LDSS
  unsigned short* sP  = (unsigned short*)(smem + OFF_P);    // 64 x LDP

  const int tid = threadIdx.x, lane = tid & 31, wave = tid >> 5;
  const int wid = blockIdx.x;
  const int b = wid >> 8, wloc = wid & 255, wy = wloc >> 4, wx = wloc & 15;
  const long long xbase = (long long)b * (128 * 128);

  // warm the L2/WGP$ path for the first weight slabs (global_prefetch_b8)
  __builtin_prefetch(wqkvT + (size_t)tid * 128, 0, 1);

  // Stage 0: gather 8x8 window (64 tokens x 384 ch), f32 -> bf16 LDS (float4)
  for (int e = tid; e < 64 * 96; e += NTHREADS) {
    const int t = e / 96, c = (e - t * 96) * 4;
    const int row = wy * 8 + (t >> 3), col = wx * 8 + (t & 7);
    const long long gi = (xbase + (long long)row * 128 + col) * 384 + c;
    const float4 v = *reinterpret_cast<const float4*>(x + gi);
    unsigned int* dst = reinterpret_cast<unsigned int*>(&sX[t * LDX + c]);
    dst[0] = pack2bf(v.x, v.y);
    dst[1] = pack2bf(v.z, v.w);
  }
  __syncthreads();

  // Stage 1: [Q|K|V] = Xw @ Wqkv + b   (64x1152; 4 m-tiles x 36 n-tile-pairs)
  // m0 = job&3 is wave-invariant (stride 16), so every job of a wave reloads the same A rows from LDS.
  for (int job = wave; job < 144; job += NWAVES) {
    const int m0 = (job & 3) * 16, n0 = (job >> 2) * 32;
    v8f acc0 = {}, acc1 = {};
#pragma unroll
    for (int k0 = 0; k0 < 384; k0 += 32) {
      v16bf a  = loadA(sX, LDX, m0, k0, lane);
      v16bf b0 = loadB(wqkvT, 384, n0,      k0, lane);
      v16bf b1 = loadB(wqkvT, 384, n0 + 16, k0, lane);
      acc0 = bfmma(a, b0, acc0);
      acc1 = bfmma(a, b1, acc1);
    }
    const int g = lane >> 4, n = lane & 15;
#pragma unroll
    for (int half = 0; half < 2; ++half) {
      const v8f acc = half ? acc1 : acc0;
      const int j = n0 + 16 * half + n;
      const float bias = b_qkv[j];
#pragma unroll
      for (int r = 0; r < 8; ++r) {
        const int m = m0 + r + 8 * g;
        const float v = acc[r] + bias;
        if (j < 384)      sQ[m * LDX + j]           = f2bf(v);
        else if (j < 768) sK[m * LDX + (j - 384)]   = f2bf(v);
        else              sVT[(j - 768) * LDVT + m] = f2bf(v); // transposed
      }
    }
  }
  __syncthreads();

  // Stage 2: S = Q @ K^T  (64x64; one 16x16 tile per wave, 16 waves = 16 tiles)
  {
    const int m0 = (wave & 3) * 16, n0 = (wave >> 2) * 16;
    v8f acc0 = {};
#pragma unroll
    for (int k0 = 0; k0 < 384; k0 += 32) {
      v16bf a  = loadA(sQ, LDX, m0, k0, lane);
      v16bf b0 = loadB(sK, LDX, n0, k0, lane);
      acc0 = bfmma(a, b0, acc0);
    }
    const int g = lane >> 4, n = lane & 15;
#pragma unroll
    for (int r = 0; r < 8; ++r)
      sS[(m0 + r + 8 * g) * LDSS + n0 + n] = acc0[r];
  }
  __syncthreads();

  // Softmax: 8 threads per row (512 threads <-> 64 rows), shfl_xor reductions
  // inside 8-lane groups. Reference has NO 1/sqrt(d) scaling.
  {
    const int row = tid >> 3, sub = tid & 7;
    float v[8];
    float mx = -3.4e38f;
#pragma unroll
    for (int i = 0; i < 8; ++i) {
      v[i] = sS[row * LDSS + sub * 8 + i];
      mx = fmaxf(mx, v[i]);
    }
#pragma unroll
    for (int d = 1; d < 8; d <<= 1) mx = fmaxf(mx, __shfl_xor(mx, d, 32));
    float sum = 0.f;
#pragma unroll
    for (int i = 0; i < 8; ++i) { v[i] = __expf(v[i] - mx); sum += v[i]; }
#pragma unroll
    for (int d = 1; d < 8; d <<= 1) sum += __shfl_xor(sum, d, 32);
    const float inv = 1.f / sum;
    unsigned int* dst = reinterpret_cast<unsigned int*>(&sP[row * LDP + sub * 8]);
#pragma unroll
    for (int i = 0; i < 4; ++i)
      dst[i] = pack2bf(v[2 * i] * inv, v[2 * i + 1] * inv);
  }
  __syncthreads();

  // Stage 3: O = P @ V  (64x384; K=64), result bf16 into sX (reuse)
  for (int job = wave; job < 48; job += NWAVES) {
    const int m0 = (job & 3) * 16, n0 = (job >> 2) * 32;
    v8f acc0 = {}, acc1 = {};
#pragma unroll
    for (int k0 = 0; k0 < 64; k0 += 32) {
      v16bf a  = loadA(sP, LDP, m0, k0, lane);
      v16bf b0 = loadB(sVT, LDVT, n0,      k0, lane);
      v16bf b1 = loadB(sVT, LDVT, n0 + 16, k0, lane);
      acc0 = bfmma(a, b0, acc0);
      acc1 = bfmma(a, b1, acc1);
    }
    const int g = lane >> 4, n = lane & 15;
#pragma unroll
    for (int r = 0; r < 8; ++r) {
      const int m = m0 + r + 8 * g;
      sX[m * LDX + n0 + n]      = f2bf(acc0[r]);
      sX[m * LDX + n0 + 16 + n] = f2bf(acc1[r]);
    }
  }
  __syncthreads();

  // Stage 4: Y = O @ Wproj + b, scatter f32 to global (B, H*W, C) layout
  for (int job = wave; job < 48; job += NWAVES) {
    const int m0 = (job & 3) * 16, n0 = (job >> 2) * 32;
    v8f acc0 = {}, acc1 = {};
#pragma unroll
    for (int k0 = 0; k0 < 384; k0 += 32) {
      v16bf a  = loadA(sX, LDX, m0, k0, lane);
      v16bf b0 = loadB(wprojT, 384, n0,      k0, lane);
      v16bf b1 = loadB(wprojT, 384, n0 + 16, k0, lane);
      acc0 = bfmma(a, b0, acc0);
      acc1 = bfmma(a, b1, acc1);
    }
    const int g = lane >> 4, n = lane & 15;
#pragma unroll
    for (int half = 0; half < 2; ++half) {
      const v8f acc = half ? acc1 : acc0;
      const int j = n0 + 16 * half + n;
      const float bias = b_proj[j];
#pragma unroll
      for (int r = 0; r < 8; ++r) {
        const int t = m0 + r + 8 * g;
        const int row = wy * 8 + (t >> 3), col = wx * 8 + (t & 7);
        const long long gi = (xbase + (long long)row * 128 + col) * 384 + j;
        out[gi] = acc[r] + bias;
      }
    }
  }
}

extern "C" void kernel_launch(void* const* d_in, const int* in_sizes, int n_in,
                              void* d_out, int out_size, void* d_ws, size_t ws_size,
                              hipStream_t stream) {
  const float* x      = (const float*)d_in[0];
  const float* w_qkv  = (const float*)d_in[1];
  const float* b_qkv  = (const float*)d_in[2];
  const float* w_proj = (const float*)d_in[3];
  const float* b_proj = (const float*)d_in[4];
  // d_in[5]/d_in[6] are h=128, w=128 (fixed by problem size)

  unsigned short* wqkvT  = (unsigned short*)d_ws;
  unsigned short* wprojT =
      (unsigned short*)((char*)d_ws + (size_t)1152 * 384 * sizeof(unsigned short));

  const int B = in_sizes[0] / (128 * 128 * 384);

  const int prepTotal = 1152 * 384 + 384 * 384;
  prep_weights_kernel<<<(prepTotal + 255) / 256, 256, 0, stream>>>(
      w_qkv, w_proj, wqkvT, wprojT);

  win_attn_kernel<<<dim3(B * 256), dim3(NTHREADS), 229376, stream>>>(
      x, b_qkv, b_proj, wqkvT, wprojT, (float*)d_out);
}
